// Self_Attention_17575006175806
// MI455X (gfx1250) — matrix operand order repaired and checked by reading, not compiled, not run on hardware
//
#include <hip/hip_runtime.h>

typedef __bf16 bf16;
typedef __attribute__((ext_vector_type(16))) __bf16 v16bf;
typedef __attribute__((ext_vector_type(8)))  __bf16 v8bf;
typedef __attribute__((ext_vector_type(8)))  float  v8f;
typedef __attribute__((ext_vector_type(4)))  float  v4f;

#define BN   8
#define CH   256
#define CQ   32
#define NPIX 4096

// ---------------------------------------------------------------------------
// Kernel A: fused projections.  W_all = [f_w; g_w; h_w]  (320 x 256)
//   F,G -> [b][n][32]  bf16   (pixel-major, 32 channels contiguous)
//   H   -> [b][c][n]   bf16   (channel-major, like input)
// X tile (32c x 16n) staged to LDS as [n][c] so the B-operand per lane is a
// pair of contiguous 8-element bf16 runs.
// ---------------------------------------------------------------------------
__global__ __launch_bounds__(128)
void proj_kernel(const float* __restrict__ x,
                 const float* __restrict__ fw, const float* __restrict__ fbias,
                 const float* __restrict__ gw, const float* __restrict__ gbias,
                 const float* __restrict__ hw, const float* __restrict__ hbias,
                 bf16* __restrict__ Fb, bf16* __restrict__ Gb, bf16* __restrict__ Hb)
{
    __shared__ bf16 Xs[16][32];              // [n][c]

    const int blk    = blockIdx.x;           // b*256 + ntile
    const int b      = blk >> 8;
    const int nbase  = (blk & 255) * 16;
    const int tid    = threadIdx.x;
    const int lid    = tid & 31;
    const int w      = tid >> 5;             // wave id 0..3
    const int lane16 = lid & 15;
    const int hi     = lid >> 4;             // 0 or 1
    const int base0  = hi ? 8 : 0;

    v8f acc[5] = {};
    const float* xb = x + (size_t)b * CH * NPIX;

    for (int ks = 0; ks < 8; ++ks) {
        const int kbase = ks * 32;
        __syncthreads();                     // protect Xs vs previous reads
        #pragma unroll
        for (int e = 0; e < 4; ++e) {
            const int n = tid & 15;
            const int c = (tid >> 4) + e * 8;        // 0..31
            const float v = xb[(size_t)(kbase + c) * NPIX + nbase + n];
            Xs[n][c] = (bf16)v;
        }
        __syncthreads();

        // B operand: lane holds pixel n = lane16, K = hi?{8..15,24..31}:{0..7,16..23}
        v16bf bop;
        {
            const bf16* p = &Xs[lane16][hi ? 8 : 0];
            v8bf b0 = *(const v8bf*)p;
            v8bf b1 = *(const v8bf*)(p + 16);
            #pragma unroll
            for (int v = 0; v < 8; ++v) { bop[v] = b0[v]; bop[v + 8] = b1[v]; }
        }

        #pragma unroll
        for (int t = 0; t < 5; ++t) {
            const int mt = w * 5 + t;                // 0..19
            const int m  = mt * 16 + lane16;         // stacked output row 0..319
            const float* wrow;
            if (m < 32)      wrow = fw + (size_t)m * CH;
            else if (m < 64) wrow = gw + (size_t)(m - 32) * CH;
            else             wrow = hw + (size_t)(m - 64) * CH;
            v4f a0 = *(const v4f*)(wrow + kbase + base0);
            v4f a1 = *(const v4f*)(wrow + kbase + base0 + 4);
            v4f a2 = *(const v4f*)(wrow + kbase + 16 + base0);
            v4f a3 = *(const v4f*)(wrow + kbase + 16 + base0 + 4);
            v16bf aop;
            #pragma unroll
            for (int v = 0; v < 4; ++v) {
                aop[v]      = (bf16)a0[v];
                aop[v + 4]  = (bf16)a1[v];
                aop[v + 8]  = (bf16)a2[v];
                aop[v + 12] = (bf16)a3[v];
            }
            acc[t] = __builtin_amdgcn_wmma_f32_16x16x32_bf16(
                false, aop, false, bop, (short)0, acc[t], false, false);
        }
    }

    const int npix = nbase + lane16;
    #pragma unroll
    for (int t = 0; t < 5; ++t) {
        const int mt = w * 5 + t;
        #pragma unroll
        for (int r = 0; r < 8; ++r) {
            const int m = mt * 16 + (hi ? 8 + r : r);
            float v = acc[t][r];
            if (m < 32) {
                v += fbias[m];
                Fb[((size_t)b * NPIX + npix) * CQ + m] = (bf16)v;
            } else if (m < 64) {
                v += gbias[m - 32];
                Gb[((size_t)b * NPIX + npix) * CQ + (m - 32)] = (bf16)v;
            } else {
                v += hbias[m - 64];
                Hb[((size_t)b * CH + (m - 64)) * NPIX + npix] = (bf16)v;
            }
        }
    }
}

// ---------------------------------------------------------------------------
// Kernel B: per-column softmax stats.  One wave per (b, 2 j-tiles of 16).
// Each F tile (A operand) is loaded once and feeds 2 WMMAs.
// ---------------------------------------------------------------------------
__global__ __launch_bounds__(32)
void stats_kernel(const bf16* __restrict__ Fb, const bf16* __restrict__ Gb,
                  float* __restrict__ Mj, float* __restrict__ Rj)
{
    const int blk    = blockIdx.x;           // b*128 + jpair
    const int b      = blk >> 7;
    const int jbase0 = (blk & 127) * 32;
    const int lid    = threadIdx.x & 31;
    const int lane16 = lid & 15;
    const int hi     = lid >> 4;

    v16bf bop[2];
    #pragma unroll
    for (int q = 0; q < 2; ++q) {
        const bf16* p = Gb + ((size_t)b * NPIX + jbase0 + q * 16 + lane16) * CQ + (hi ? 8 : 0);
        v8bf b0 = *(const v8bf*)p;
        v8bf b1 = *(const v8bf*)(p + 16);
        #pragma unroll
        for (int v = 0; v < 8; ++v) { bop[q][v] = b0[v]; bop[q][v + 8] = b1[v]; }
    }

    float m[2] = { -3.0e38f, -3.0e38f };
    float s[2] = { 0.0f, 0.0f };

    for (int it = 0; it < 256; ++it) {
        const bf16* p = Fb + ((size_t)b * NPIX + it * 16 + lane16) * CQ;
        v8bf a0 = *(const v8bf*)(p + (hi ? 8 : 0));
        v8bf a1 = *(const v8bf*)(p + 16 + (hi ? 8 : 0));
        v16bf aop;
        #pragma unroll
        for (int v = 0; v < 8; ++v) { aop[v] = a0[v]; aop[v + 8] = a1[v]; }

        #pragma unroll
        for (int q = 0; q < 2; ++q) {
            v8f d = {};
            d = __builtin_amdgcn_wmma_f32_16x16x32_bf16(
                false, aop, false, bop[q], (short)0, d, false, false);
            float tm = d[0];
            #pragma unroll
            for (int r = 1; r < 8; ++r) tm = fmaxf(tm, d[r]);
            const float mn = fmaxf(m[q], tm);
            const float sc = __expf(m[q] - mn);
            float part = 0.0f;
            #pragma unroll
            for (int r = 0; r < 8; ++r) part += __expf(d[r] - mn);
            s[q] = s[q] * sc + part;
            m[q] = mn;
        }
    }
    #pragma unroll
    for (int q = 0; q < 2; ++q) {
        const float mo = __shfl_xor(m[q], 16, 32);
        const float so = __shfl_xor(s[q], 16, 32);
        const float mf = fmaxf(m[q], mo);
        const float sf = s[q] * __expf(m[q] - mf) + so * __expf(mo - mf);
        if (lid < 16) {
            Mj[(size_t)b * NPIX + jbase0 + q * 16 + lane16] = mf;
            Rj[(size_t)b * NPIX + jbase0 + q * 16 + lane16] = 1.0f / sf;
        }
    }
}

// ---------------------------------------------------------------------------
// Kernel C: output pass.  One wave per (b, 2 i-tiles, c-half[128 channels]).
// Per 32-j step: 4 S-WMMAs (G tile shared across both i-tiles), P scaled by
// per-column exp stats, LDS transpose to A layout, then 8 H tiles each feed
// 2 accumulated WMMAs.  20 WMMAs per step against one set of B operands.
// ---------------------------------------------------------------------------
__global__ __launch_bounds__(32)
void attn_out_kernel(const bf16* __restrict__ Fb, const bf16* __restrict__ Gb,
                     const bf16* __restrict__ Hb,
                     const float* __restrict__ Mj, const float* __restrict__ Rj,
                     const float* __restrict__ x, float* __restrict__ out)
{
    __shared__ bf16 Ps[2][16][32];           // [itile][i_local][j_local]

    const int blk    = blockIdx.x;           // ((b*128 + ipair)*2 + chalf)
    const int chalf  = blk & 1;
    const int ipair  = (blk >> 1) & 127;
    const int b      = blk >> 8;
    const int ibase0 = ipair * 32;
    const int lid    = threadIdx.x & 31;
    const int lane16 = lid & 15;
    const int hi     = lid >> 4;

    // A operands for S: F^T tiles at ibase0, ibase0+16 (loop invariant)
    v16bf aF[2];
    #pragma unroll
    for (int t = 0; t < 2; ++t) {
        const bf16* p = Fb + ((size_t)b * NPIX + ibase0 + t * 16 + lane16) * CQ;
        v8bf a0 = *(const v8bf*)(p + (hi ? 8 : 0));
        v8bf a1 = *(const v8bf*)(p + 16 + (hi ? 8 : 0));
        #pragma unroll
        for (int v = 0; v < 8; ++v) { aF[t][v] = a0[v]; aF[t][v + 8] = a1[v]; }
    }

    v8f acc0[8] = {};
    v8f acc1[8] = {};
    const bf16* HbB = Hb + (size_t)b * CH * NPIX;

    for (int jb = 0; jb < 128; ++jb) {
        const int j0 = jb * 32;
        __builtin_prefetch(HbB + (size_t)(chalf * 128 + lane16) * NPIX + j0 + 64, 0, 0);
        __syncthreads();                     // Ps writes vs previous reads
        #pragma unroll
        for (int sub = 0; sub < 2; ++sub) {
            const int jt = j0 + sub * 16;
            v16bf bG;
            {
                const bf16* p = Gb + ((size_t)b * NPIX + jt + lane16) * CQ + (hi ? 8 : 0);
                v8bf b0 = *(const v8bf*)p;
                v8bf b1 = *(const v8bf*)(p + 16);
                #pragma unroll
                for (int v = 0; v < 8; ++v) { bG[v] = b0[v]; bG[v + 8] = b1[v]; }
            }
            const int j  = jt + lane16;
            const float mj = Mj[(size_t)b * NPIX + j];
            const float rj = Rj[(size_t)b * NPIX + j];
            #pragma unroll
            for (int t = 0; t < 2; ++t) {
                v8f d = {};
                d = __builtin_amdgcn_wmma_f32_16x16x32_bf16(
                    false, aF[t], false, bG, (short)0, d, false, false);
                #pragma unroll
                for (int r = 0; r < 8; ++r) {
                    const float pv = __expf(d[r] - mj) * rj;
                    Ps[t][hi ? 8 + r : r][sub * 16 + lane16] = (bf16)pv;
                }
            }
        }
        __syncthreads();

        // re-read P blocks in A-operand layout (row i per lane)
        v16bf pA[2];
        #pragma unroll
        for (int t = 0; t < 2; ++t) {
            const bf16* p = &Ps[t][lane16][hi ? 16 : 0];
            v8bf a0 = *(const v8bf*)p;
            v8bf a1 = *(const v8bf*)(p + 8);
            #pragma unroll
            for (int v = 0; v < 8; ++v) { pA[t][v] = a0[v]; pA[t][v + 8] = a1[v]; }
        }

        #pragma unroll
        for (int ct = 0; ct < 8; ++ct) {
            const int c = (chalf * 8 + ct) * 16 + lane16;
            const bf16* p = HbB + (size_t)c * NPIX + j0 + (hi ? 16 : 0);
            v8bf h0 = *(const v8bf*)p;
            v8bf h1 = *(const v8bf*)(p + 8);
            v16bf bH;
            #pragma unroll
            for (int v = 0; v < 8; ++v) { bH[v] = h0[v]; bH[v + 8] = h1[v]; }
            acc0[ct] = __builtin_amdgcn_wmma_f32_16x16x32_bf16(
                false, pA[0], false, bH, (short)0, acc0[ct], false, false);
            acc1[ct] = __builtin_amdgcn_wmma_f32_16x16x32_bf16(
                false, pA[1], false, bH, (short)0, acc1[ct], false, false);
        }
    }

    // epilogue: out = O + x  (GAMMA == 1.0)
    #pragma unroll
    for (int ct = 0; ct < 8; ++ct) {
        const int c = (chalf * 8 + ct) * 16 + lane16;
        #pragma unroll
        for (int r = 0; r < 8; ++r) {
            const int m = hi ? 8 + r : r;
            {
                const size_t idx = ((size_t)b * CH + c) * NPIX + (ibase0 + m);
                out[idx] = acc0[ct][r] + x[idx];
            }
            {
                const size_t idx = ((size_t)b * CH + c) * NPIX + (ibase0 + 16 + m);
                out[idx] = acc1[ct][r] + x[idx];
            }
        }
    }
}

// ---------------------------------------------------------------------------
extern "C" void kernel_launch(void* const* d_in, const int* in_sizes, int n_in,
                              void* d_out, int out_size, void* d_ws, size_t ws_size,
                              hipStream_t stream)
{
    const float* x  = (const float*)d_in[0];
    const float* fw = (const float*)d_in[1];
    const float* fb = (const float*)d_in[2];
    const float* gw = (const float*)d_in[3];
    const float* gb = (const float*)d_in[4];
    const float* hw = (const float*)d_in[5];
    const float* hb = (const float*)d_in[6];
    float* out = (float*)d_out;

    char* ws = (char*)d_ws;
    bf16*  Fb = (bf16*)(ws);                                   //  2 MB
    bf16*  Gb = (bf16*)(ws + (size_t)2  * 1024 * 1024);        //  2 MB
    bf16*  Hb = (bf16*)(ws + (size_t)4  * 1024 * 1024);        // 16 MB
    float* Mj = (float*)(ws + (size_t)20 * 1024 * 1024);       // 128 KB
    float* Rj = (float*)(ws + (size_t)20 * 1024 * 1024 + 131072);

    proj_kernel<<<BN * (NPIX / 16), 128, 0, stream>>>(x, fw, fb, gw, gb, hw, hb,
                                                      Fb, Gb, Hb);
    stats_kernel<<<BN * (NPIX / 32), 32, 0, stream>>>(Fb, Gb, Mj, Rj);
    attn_out_kernel<<<BN * (NPIX / 32) * 2, 32, 0, stream>>>(Fb, Gb, Hb, Mj, Rj,
                                                             x, out);
}
